// PerLevelMDVSCV2_60327110640101
// MI455X (gfx1250) — matrix operand order repaired and checked by reading, not compiled, not run on hardware
//
#include <hip/hip_runtime.h>
#include <math.h>

// ---------------------------------------------------------------------------
// Problem constants (match reference)
// ---------------------------------------------------------------------------
#define B_SZ   2
#define T_SZ   4
#define NIMG   8          // B*T
#define INCH   128
#define MIDC   96
#define LATC   64
#define H_DIM  128
#define W_DIM  128
#define HW     16384      // 128*128, == 1<<14

#define FLAG_GELU_IN  1
#define FLAG_GELU_OUT 2

typedef unsigned short bfu;   // raw bf16 bits in storage

// ---------------------------------------------------------------------------
// Types for WMMA
// ---------------------------------------------------------------------------
typedef __attribute__((ext_vector_type(16))) __bf16    bf16x16;
typedef __attribute__((ext_vector_type(8)))  float     f32x8;
typedef __attribute__((ext_vector_type(4)))  unsigned  u32x4;

union FragU { u32x4 q[2]; bf16x16 v; };

__device__ __forceinline__ unsigned f2bfu(float f) {
    unsigned u = __float_as_uint(f);
    unsigned r = u + 0x7FFFu + ((u >> 16) & 1u);   // round-to-nearest-even
    return r >> 16;
}
__device__ __forceinline__ bfu   f2bf(float f) { return (bfu)f2bfu(f); }
__device__ __forceinline__ float bf2f(bfu h)   { return __uint_as_float(((unsigned)h) << 16); }

__device__ __forceinline__ float gelu_f(float x) {
    return 0.5f * x * (1.0f + erff(x * 0.70710678118654752f));
}

// ---------------------------------------------------------------------------
// 1x1 conv == GEMM over spatial positions, bf16 WMMA, fp32 accumulate.
//   out[img, co, p] = act( sum_k W[co,k] * X[img,k,p] + bias[co] ) (+ resid)
// X = concat(x1: c1 ch, x2: c2 ch); c1 is a multiple of the K-step (32), so
// the x1/x2 switch is a scalar-uniform per-iteration branch (single pass).
// INF32: activations fp32 (network input); else bf16-resident.
// OUTF32: write fp32 (d_out score tensors); else bf16.
// Tile: 32M x 128N, 8 waves (2Mx4N), each wave = two 16x16 WMMA accumulators.
// ---------------------------------------------------------------------------
template<bool INF32, bool OUTF32>
__global__ __launch_bounds__(256)
void conv1x1_wmma_kernel(const void* __restrict__ x1_, int c1,
                         const void* __restrict__ x2_, int c2,
                         const float* __restrict__ w,
                         const float* __restrict__ bias,
                         const bfu* __restrict__ resid,
                         void* __restrict__ out_,
                         int Co, int flags)
{
    __shared__ __attribute__((aligned(16))) unsigned short As[32][40];  // [M][K] rows 80B
    __shared__ __attribute__((aligned(16))) unsigned short Bs[128][40]; // [N][K] rows 80B

    const int Cin = c1 + c2;
    const int p0  = blockIdx.x * 128;
    const int co0 = blockIdx.y * 32;
    const size_t img = blockIdx.z;

    const float* xf1 = (const float*)x1_;
    const float* xf2 = (const float*)x2_;
    const bfu*   xh1 = (const bfu*)x1_;
    const bfu*   xh2 = (const bfu*)x2_;
    float* of = (float*)out_;
    bfu*   oh = (bfu*)out_;
    const size_t oOff = img * (size_t)Co * HW;
    const bfu* ri = resid ? (resid + img * (size_t)Co * HW) : nullptr;

    const int tid  = threadIdx.x;
    const int lane = tid & 31;
    const int wave = tid >> 5;
    const int m0 = (wave & 1) << 4;   // 2 waves along M (16 each)
    const int n0 = (wave >> 1) << 5;  // 4 waves along N (32 each)
    const int lr = lane & 15;
    const int hi = lane >> 4;         // 0 or 1

    // staging maps
    const int a_m  = tid >> 2;           // rows 0..31 (only tid<128 stage A)
    const int a_kq = (tid & 3) << 3;     // k chunk of 8
    const int b_n  = tid & 127;          // fp32 path: 128 columns
    const int b_kq = (tid >> 7) << 3;    // fp32 path: 0 or 8 (+16 second chunk)
    const int b_n2  = (tid & 63) << 1;   // bf16 path: even column (pair n2,n2+1)
    const int b_kq8 = (tid >> 6) << 3;   // bf16 path: 0/8/16/24

    const bool aRowsOk = (co0 + 32 <= Co);

    f32x8 acc0 = {0.f,0.f,0.f,0.f,0.f,0.f,0.f,0.f};
    f32x8 acc1 = {0.f,0.f,0.f,0.f,0.f,0.f,0.f,0.f};

    for (int k0 = 0; k0 < Cin; k0 += 32) {
        const bool useX1 = (k0 < c1);
        const int  kbase = useX1 ? k0 : (k0 - c1);
        const int  lim   = useX1 ? c1 : c2;
        const size_t xoff = img * (size_t)lim * HW;   // lim == that input's channel count
        const bool kOkB = (kbase + 32 <= lim);
        const bool kOkA = (k0 + 32 <= Cin);

        // ---- stage A (weights, fp32): waves 0..3, one ds_store_b128 each ----
        if (tid < 128) {
            unsigned d[4];
            if (aRowsOk && kOkA) {
                const float* wr = w + (size_t)(co0 + a_m) * Cin + k0 + a_kq;
                #pragma unroll
                for (int j = 0; j < 4; ++j)
                    d[j] = f2bfu(wr[2*j]) | (f2bfu(wr[2*j + 1]) << 16);
            } else {
                int gm  = co0 + a_m;
                int gmc = (gm < Co) ? gm : (Co - 1);
                const float* wr = w + (size_t)gmc * Cin;
                #pragma unroll
                for (int j = 0; j < 4; ++j) {
                    int k_lo = k0 + a_kq + 2*j, k_hi = k_lo + 1;
                    int kcl = (k_lo < Cin) ? k_lo : (Cin - 1);
                    int kch = (k_hi < Cin) ? k_hi : (Cin - 1);
                    float vlo = wr[kcl];
                    float vhi = wr[kch];
                    if (gm >= Co || k_lo >= Cin) vlo = 0.f;
                    if (gm >= Co || k_hi >= Cin) vhi = 0.f;
                    d[j] = f2bfu(vlo) | (f2bfu(vhi) << 16);
                }
            }
            u32x4 q = { d[0], d[1], d[2], d[3] };
            *(u32x4*)&As[a_m][a_kq] = q;
        }

        // ---- stage B (activations) ----
        if (INF32) {
            const float* xc = (useX1 ? xf1 : xf2) + xoff;
            #pragma unroll
            for (int cch = 0; cch < 2; ++cch) {
                const int kq = b_kq + (cch << 4);   // 0/8 then 16/24
                unsigned d[4];
                if (kOkB) {
                    const float* xr = xc + (size_t)(kbase + kq) * HW + p0 + b_n;
                    #pragma unroll
                    for (int j = 0; j < 4; ++j) {
                        float vlo = xr[(size_t)(2*j)     * HW];
                        float vhi = xr[(size_t)(2*j + 1) * HW];
                        if (flags & FLAG_GELU_IN) { vlo = gelu_f(vlo); vhi = gelu_f(vhi); }
                        d[j] = f2bfu(vlo) | (f2bfu(vhi) << 16);
                    }
                } else {
                    #pragma unroll
                    for (int j = 0; j < 4; ++j) {
                        int k_lo = kbase + kq + 2*j, k_hi = k_lo + 1;
                        int kcl = (k_lo < lim) ? k_lo : (lim - 1);
                        int kch = (k_hi < lim) ? k_hi : (lim - 1);
                        float vlo = xc[(size_t)kcl * HW + p0 + b_n];
                        float vhi = xc[(size_t)kch * HW + p0 + b_n];
                        if (k_lo >= lim) vlo = 0.f;
                        if (k_hi >= lim) vhi = 0.f;
                        if (flags & FLAG_GELU_IN) { vlo = gelu_f(vlo); vhi = gelu_f(vhi); }
                        d[j] = f2bfu(vlo) | (f2bfu(vhi) << 16);
                    }
                }
                u32x4 q = { d[0], d[1], d[2], d[3] };
                *(u32x4*)&Bs[b_n][kq] = q;
            }
            if (kbase + 32 < lim)
                __builtin_prefetch(xc + (size_t)(kbase + 32 + b_kq) * HW + p0 + b_n, 0, 1);
        } else {
            // bf16 activations: one u32 = two adjacent columns; no conversion.
            const bfu* xc = (useX1 ? xh1 : xh2) + xoff;
            bfu lo8[8], hi8[8];
            if (kOkB) {
                const bfu* xr = xc + (size_t)(kbase + b_kq8) * HW + p0 + b_n2;
                #pragma unroll
                for (int j = 0; j < 8; ++j) {
                    unsigned v = *(const unsigned*)(xr + (size_t)j * HW);
                    bfu l = (bfu)(v & 0xFFFFu), h2 = (bfu)(v >> 16);
                    if (flags & FLAG_GELU_IN) { l = f2bf(gelu_f(bf2f(l))); h2 = f2bf(gelu_f(bf2f(h2))); }
                    lo8[j] = l; hi8[j] = h2;
                }
            } else {
                #pragma unroll
                for (int j = 0; j < 8; ++j) {
                    int kl  = kbase + b_kq8 + j;
                    int klc = (kl < lim) ? kl : (lim - 1);
                    unsigned v = *(const unsigned*)(xc + (size_t)klc * HW + p0 + b_n2);
                    bfu l = (bfu)(v & 0xFFFFu), h2 = (bfu)(v >> 16);
                    if (kl >= lim) { l = 0; h2 = 0; }
                    if (flags & FLAG_GELU_IN) { l = f2bf(gelu_f(bf2f(l))); h2 = f2bf(gelu_f(bf2f(h2))); }
                    lo8[j] = l; hi8[j] = h2;
                }
            }
            u32x4 qlo = { (unsigned)lo8[0] | ((unsigned)lo8[1] << 16),
                          (unsigned)lo8[2] | ((unsigned)lo8[3] << 16),
                          (unsigned)lo8[4] | ((unsigned)lo8[5] << 16),
                          (unsigned)lo8[6] | ((unsigned)lo8[7] << 16) };
            u32x4 qhi = { (unsigned)hi8[0] | ((unsigned)hi8[1] << 16),
                          (unsigned)hi8[2] | ((unsigned)hi8[3] << 16),
                          (unsigned)hi8[4] | ((unsigned)hi8[5] << 16),
                          (unsigned)hi8[6] | ((unsigned)hi8[7] << 16) };
            *(u32x4*)&Bs[b_n2][b_kq8]     = qlo;
            *(u32x4*)&Bs[b_n2 + 1][b_kq8] = qhi;
            if (kbase + 32 < lim)
                __builtin_prefetch(xc + (size_t)(kbase + 32 + b_kq8) * HW + p0 + b_n2, 0, 1);
        }

        __syncthreads();

        // ---- fragments per ISA layouts ----
        FragU ua, ub0, ub1;
        // A 16x32: lane<16 -> K {0..7,16..23}; lane>=16 -> K {8..15,24..31}
        ua.q[0] = *(const u32x4*)&As[m0 + lr][hi * 8];
        ua.q[1] = *(const u32x4*)&As[m0 + lr][hi * 8 + 16];
        // B 32x16: lane<16 -> K {0..7,16..23}; lane>=16 -> K {8..15,24..31} (column N=lr)
        ub0.q[0] = *(const u32x4*)&Bs[n0 + lr][hi * 8];
        ub0.q[1] = *(const u32x4*)&Bs[n0 + lr][hi * 8 + 16];
        ub1.q[0] = *(const u32x4*)&Bs[n0 + 16 + lr][hi * 8];
        ub1.q[1] = *(const u32x4*)&Bs[n0 + 16 + lr][hi * 8 + 16];

        acc0 = __builtin_amdgcn_wmma_f32_16x16x32_bf16(
                 false, ua.v, false, ub0.v, (short)0, acc0, false, false);
        acc1 = __builtin_amdgcn_wmma_f32_16x16x32_bf16(
                 false, ua.v, false, ub1.v, (short)0, acc1, false, false);
        __syncthreads();
    }

    // ---- epilogue: C layout vgpr r -> M = r + 8*hi, N = lr ----
    #pragma unroll
    for (int r = 0; r < 8; ++r) {
        int gm = co0 + m0 + (hi << 3) + r;
        if (gm < Co) {
            size_t row = oOff + (size_t)gm * HW;
            int gp0 = p0 + n0 + lr;
            int gp1 = gp0 + 16;
            float v0 = acc0[r] + bias[gm];
            float v1 = acc1[r] + bias[gm];
            if (flags & FLAG_GELU_OUT) { v0 = gelu_f(v0); v1 = gelu_f(v1); }
            if (ri) {
                size_t rrow = (size_t)gm * HW;
                v0 += bf2f(ri[rrow + gp0]);
                v1 += bf2f(ri[rrow + gp1]);
            }
            if (OUTF32) { of[row + gp0] = v0;      of[row + gp1] = v1; }
            else        { oh[row + gp0] = f2bf(v0); oh[row + gp1] = f2bf(v1); }
        }
    }
}

// ---------------------------------------------------------------------------
// GroupNorm (groups=8) + exact GELU (bf16 in/out), block per (image, group)
// ---------------------------------------------------------------------------
__global__ __launch_bounds__(256)
void gn_gelu_kernel(const bfu* __restrict__ x,
                    const float* __restrict__ gw, const float* __restrict__ gb,
                    bfu* __restrict__ out, int C)
{
    const int bx = blockIdx.x, n = bx >> 3, g = bx & 7;
    const int cpg = C >> 3;
    const size_t base = ((size_t)n * C + (size_t)g * cpg) * HW;
    const bfu* xi = x + base;
    bfu*       oi = out + base;
    const int cnt = cpg * HW;

    float s = 0.f, s2 = 0.f;
    for (int i = threadIdx.x; i < cnt; i += 256) { float v = bf2f(xi[i]); s += v; s2 += v*v; }
    __shared__ float r1[256], r2[256];
    r1[threadIdx.x] = s; r2[threadIdx.x] = s2; __syncthreads();
    for (int st = 128; st > 0; st >>= 1) {
        if (threadIdx.x < st) { r1[threadIdx.x] += r1[threadIdx.x+st]; r2[threadIdx.x] += r2[threadIdx.x+st]; }
        __syncthreads();
    }
    const float ic = 1.f / (float)cnt;
    const float mean = r1[0] * ic;
    const float var  = r2[0] * ic - mean * mean;
    const float rstd = rsqrtf(var + 1e-5f);
    for (int i = threadIdx.x; i < cnt; i += 256) {
        int c = g * cpg + i / HW;
        float v = (bf2f(xi[i]) - mean) * rstd * gw[c] + gb[c];
        oi[i] = f2bf(gelu_f(v));
    }
}

// ---------------------------------------------------------------------------
// Grouped 3x3 conv (bf16 in/out), Cin==Cout==C, weight [C][cpg][3][3]
// ---------------------------------------------------------------------------
__global__ __launch_bounds__(256)
void conv3x3_kernel(const bfu* __restrict__ x, const float* __restrict__ w,
                    const float* __restrict__ b, bfu* __restrict__ out,
                    int C, int groups)
{
    size_t idx = (size_t)blockIdx.x * 256 + threadIdx.x;
    int p = (int)(idx & (HW - 1));
    size_t t = idx >> 14;
    int c = (int)(t % C), n = (int)(t / C);
    int h = p >> 7, ww = p & 127;
    int cpg = C / groups, g = c / cpg;
    float acc = b[c];
    for (int ic = 0; ic < cpg; ++ic) {
        const bfu* xi = x + ((size_t)n * C + g * cpg + ic) * HW;
        const float* wp = w + ((size_t)c * cpg + ic) * 9;
        #pragma unroll
        for (int ky = 0; ky < 3; ++ky) {
            int hh = h + ky - 1; if ((unsigned)hh >= 128u) continue;
            #pragma unroll
            for (int kx = 0; kx < 3; ++kx) {
                int wx = ww + kx - 1; if ((unsigned)wx >= 128u) continue;
                acc += wp[ky * 3 + kx] * bf2f(xi[hh * 128 + wx]);
            }
        }
    }
    out[idx] = f2bf(acc);
}

// ---------------------------------------------------------------------------
// SE block pieces
// ---------------------------------------------------------------------------
__global__ __launch_bounds__(256)
void se_pool_kernel(const bfu* __restrict__ x, float* __restrict__ pooled)
{
    const bfu* xi = x + (size_t)blockIdx.x * HW;
    float s = 0.f;
    for (int i = threadIdx.x; i < HW; i += 256) s += bf2f(xi[i]);
    __shared__ float r[256];
    r[threadIdx.x] = s; __syncthreads();
    for (int st = 128; st > 0; st >>= 1) {
        if (threadIdx.x < st) r[threadIdx.x] += r[threadIdx.x + st];
        __syncthreads();
    }
    if (threadIdx.x == 0) pooled[blockIdx.x] = r[0] * (1.f / (float)HW);
}

__global__ __launch_bounds__(128)
void se_mlp_kernel(const float* __restrict__ pooled,
                   const float* __restrict__ w1, const float* __restrict__ b1,
                   const float* __restrict__ w2, const float* __restrict__ b2,
                   float* __restrict__ scale, int C, int M)
{
    const int n = blockIdx.x, tid = threadIdx.x;
    __shared__ float sIn[128], sH[32];
    if (tid < C) sIn[tid] = pooled[(size_t)n * C + tid];
    __syncthreads();
    if (tid < M) {
        float a = b1[tid];
        for (int c = 0; c < C; ++c) a += w1[(size_t)tid * C + c] * sIn[c];
        sH[tid] = gelu_f(a);
    }
    __syncthreads();
    if (tid < C) {
        float a = b2[tid];
        for (int j = 0; j < M; ++j) a += w2[(size_t)tid * M + j] * sH[j];
        scale[(size_t)n * C + tid] = 1.f / (1.f + __expf(-a));
    }
}

__global__ __launch_bounds__(256)
void se_scale_kernel(const bfu* __restrict__ x, const float* __restrict__ scale,
                     bfu* __restrict__ out)
{
    size_t idx = (size_t)blockIdx.x * 256 + threadIdx.x;
    out[idx] = f2bf(bf2f(x[idx]) * scale[idx >> 14]);
}

// ---------------------------------------------------------------------------
// Temporal attention decomposition (sc/wts fp32, latent bf16)
// ---------------------------------------------------------------------------
__global__ __launch_bounds__(256)
void softmax_t_kernel(const float* __restrict__ sc, float* __restrict__ wts)
{
    size_t idx = (size_t)blockIdx.x * 256 + threadIdx.x;  // over B*HW
    int p = (int)(idx & (HW - 1)), b = (int)(idx >> 14);
    float v[T_SZ]; float mx = -1e30f;
    for (int t = 0; t < T_SZ; ++t) { v[t] = sc[((size_t)(b*T_SZ + t)) * HW + p]; mx = fmaxf(mx, v[t]); }
    float s = 0.f;
    for (int t = 0; t < T_SZ; ++t) { v[t] = __expf(v[t] - mx); s += v[t]; }
    float inv = 1.f / s;
    for (int t = 0; t < T_SZ; ++t) wts[((size_t)(b*T_SZ + t)) * HW + p] = v[t] * inv;
}

__global__ __launch_bounds__(256)
void combine_kernel(const bfu* __restrict__ z, const float* __restrict__ wts,
                    bfu* __restrict__ common, bfu* __restrict__ ind)
{
    size_t idx = (size_t)blockIdx.x * 256 + threadIdx.x;  // over B*LATC*HW
    int p = (int)(idx & (HW - 1));
    size_t t = idx >> 14;
    int c = (int)(t & 63), b = (int)(t >> 6);
    float com = 0.f, zv[T_SZ];
    for (int tt = 0; tt < T_SZ; ++tt) {
        size_t zi = (((size_t)(b*T_SZ + tt)) * LATC + c) * (size_t)HW + p;
        zv[tt] = bf2f(z[zi]);
        com += wts[((size_t)(b*T_SZ + tt)) * HW + p] * zv[tt];
    }
    common[idx] = f2bf(com);
    for (int tt = 0; tt < T_SZ; ++tt) {
        size_t zi = (((size_t)(b*T_SZ + tt)) * LATC + c) * (size_t)HW + p;
        ind[zi] = f2bf(zv[tt] - com);
    }
}

// ---------------------------------------------------------------------------
// Gate: 4x4 block pooling (scores fp32 in d_out) + radix-select top-k mask
// ---------------------------------------------------------------------------
__global__ __launch_bounds__(256)
void pool4x4_kernel(const float* __restrict__ x, float* __restrict__ pooled)
{
    size_t idx = (size_t)blockIdx.x * 256 + threadIdx.x;  // over N*C*1024
    size_t nc = idx >> 10;
    int pp = (int)(idx & 1023), ph = pp >> 5, pw = pp & 31;
    const float* xi = x + nc * HW;
    float s = 0.f;
    #pragma unroll
    for (int ky = 0; ky < 4; ++ky)
        #pragma unroll
        for (int kx = 0; kx < 4; ++kx)
            s += xi[(ph * 4 + ky) * 128 + pw * 4 + kx];
    pooled[idx] = s * (1.f / 16.f);
}

// keep the `keep` SMALLEST pooled scores per (n,c); expand 4x4 and apply mask.
__global__ __launch_bounds__(256)
void topk_mask_kernel(const float* __restrict__ pooled, const bfu* __restrict__ inp,
                      float* __restrict__ mask_out, bfu* __restrict__ prod_out, int keep)
{
    const int bc = blockIdx.x, tid = threadIdx.x;
    const float* row = pooled + (size_t)bc * 1024;
    __shared__ unsigned su[1024];
    __shared__ float smask[1024];
    __shared__ int red[256];

    for (int i = tid; i < 1024; i += 256) {
        unsigned bits = __float_as_uint(row[i]);
        su[i] = (bits & 0x80000000u) ? ~bits : (bits | 0x80000000u); // sortable uint
    }
    __syncthreads();

    unsigned prefix = 0; int need = keep;
    for (int bit = 31; bit >= 0; --bit) {
        unsigned long long hp = ((unsigned long long)prefix) >> (bit + 1);
        int c0 = 0;
        for (int i = tid; i < 1024; i += 256) {
            unsigned u = su[i];
            if (((unsigned long long)u >> (bit + 1)) == hp && !((u >> bit) & 1u)) c0++;
        }
        red[tid] = c0; __syncthreads();
        for (int st = 128; st > 0; st >>= 1) { if (tid < st) red[tid] += red[tid + st]; __syncthreads(); }
        int tot0 = red[0];
        __syncthreads();
        if (need > tot0) { need -= tot0; prefix |= (1u << bit); }
    }
    if (tid == 0) {                  // mark keeps; ties broken by index order
        int rem = need;
        for (int i = 0; i < 1024; ++i) {
            unsigned u = su[i];
            float mv = 0.f;
            if (u < prefix) mv = 1.f;
            else if (u == prefix && rem > 0) { mv = 1.f; rem--; }
            smask[i] = mv;
        }
    }
    __syncthreads();

    const bfu* ii = inp + (size_t)bc * HW;
    float* mo = mask_out + (size_t)bc * HW;
    bfu*   po = prod_out + (size_t)bc * HW;
    for (int i = tid; i < HW; i += 256) {
        int h = i >> 7, w = i & 127;
        float m = smask[(h >> 2) * 32 + (w >> 2)];
        mo[i] = m;
        po[i] = (m != 0.f) ? ii[i] : (bfu)0;
    }
}

// ---------------------------------------------------------------------------
// small glue kernels
// ---------------------------------------------------------------------------
__global__ __launch_bounds__(256)
void zdec_kernel(const bfu* __restrict__ tc, const bfu* __restrict__ ti,
                 bfu* __restrict__ zdec)
{
    size_t idx = (size_t)blockIdx.x * 256 + threadIdx.x;  // over NIMG*LATC*HW
    size_t chw = idx % ((size_t)LATC * HW);
    int n = (int)(idx / ((size_t)LATC * HW));
    int b = n >> 2;
    zdec[idx] = f2bf(bf2f(tc[(size_t)b * LATC * HW + chw]) + bf2f(ti[idx]));
}

__global__ __launch_bounds__(256)
void cvt_out_kernel(const bfu* __restrict__ a, float* __restrict__ o)
{
    size_t i = (size_t)blockIdx.x * 256 + threadIdx.x;
    o[i] = bf2f(a[i]);
}

__global__ void set_means_kernel(float* p, float a, float b) { p[0] = a; p[1] = b; }

// ---------------------------------------------------------------------------
// Host-side orchestration helpers
// ---------------------------------------------------------------------------
struct ResP  { const float *gn_w,*gn_b,*w1,*b1,*wd,*bd,*w2,*b2; };
struct GateP { const float *w1,*b1,*w3,*b3,*w2,*b2; };

static inline void launch_conv(const void* x1, int c1, const void* x2, int c2,
                               const float* w, const float* b, const bfu* resid,
                               void* out, int Co, int nimg, int flags,
                               bool inF32, bool outF32, hipStream_t s)
{
    dim3 grid(HW / 128, (Co + 31) / 32, nimg);
    if (inF32)
        conv1x1_wmma_kernel<true,false><<<grid, 256, 0, s>>>(x1, c1, x2, c2, w, b, resid, out, Co, flags);
    else if (outF32)
        conv1x1_wmma_kernel<false,true><<<grid, 256, 0, s>>>(x1, c1, x2, c2, w, b, resid, out, Co, flags);
    else
        conv1x1_wmma_kernel<false,false><<<grid, 256, 0, s>>>(x1, c1, x2, c2, w, b, resid, out, Co, flags);
}

static inline void run_res(const ResP& p, bfu* X, bfu* t1, bfu* t2,
                           int C, int nimg, hipStream_t s)
{
    gn_gelu_kernel<<<nimg * 8, 256, 0, s>>>(X, p.gn_w, p.gn_b, t1, C);
    launch_conv(t1, C, nullptr, 0, p.w1, p.b1, nullptr, t2, C, nimg, 0, false, false, s);
    conv3x3_kernel<<<(unsigned)((size_t)nimg * C * HW / 256), 256, 0, s>>>(t2, p.wd, p.bd, t1, C, C);
    launch_conv(t1, C, nullptr, 0, p.w2, p.b2, /*resid*/X, X, C, nimg, FLAG_GELU_IN, false, false, s);
}

static inline void run_se(const bfu* x, const float* w1, const float* b1,
                          const float* w2, const float* b2, bfu* out,
                          float* pool, float* scl, int C, int M, hipStream_t s)
{
    se_pool_kernel<<<NIMG * C, 256, 0, s>>>(x, pool);
    se_mlp_kernel<<<NIMG, 128, 0, s>>>(pool, w1, b1, w2, b2, scl, C, M);
    se_scale_kernel<<<(unsigned)((size_t)NIMG * C * HW / 256), 256, 0, s>>>(x, scl, out);
}

static inline void run_gate(const GateP& g, const bfu* inp, int nimg, int keep,
                            bfu* m1, bfu* m2, float* scores, float* pooled,
                            float* maskout, bfu* prodout, hipStream_t s)
{
    launch_conv(inp, LATC, nullptr, 0, g.w1, g.b1, nullptr, m1, 16, nimg, FLAG_GELU_OUT, false, false, s);
    conv3x3_kernel<<<(unsigned)((size_t)nimg * 16 * HW / 256), 256, 0, s>>>(m1, g.w3, g.b3, m2, 16, 8);
    launch_conv(m2, 16, nullptr, 0, g.w2, g.b2, nullptr, scores, LATC, nimg, FLAG_GELU_IN, false, true, s);
    pool4x4_kernel<<<(unsigned)((size_t)nimg * LATC * 1024 / 256), 256, 0, s>>>(scores, pooled);
    topk_mask_kernel<<<nimg * LATC, 256, 0, s>>>(pooled, inp, maskout, prodout, keep);
}

// ---------------------------------------------------------------------------
// kernel_launch
// ---------------------------------------------------------------------------
extern "C" void kernel_launch(void* const* d_in, const int* in_sizes, int n_in,
                              void* d_out_v, int out_size, void* d_ws, size_t ws_size,
                              hipStream_t stream)
{
    (void)in_sizes; (void)n_in; (void)out_size; (void)ws_size;
    const float* feats = (const float*)d_in[0];
    float* out = (float*)d_out_v;

    // ---- output regions (reference tuple, concatenated flat, fp32) ----
    const size_t OUT_OFF   = 0;
    const size_t CMASK_OFF = OUT_OFF   + (size_t)NIMG * INCH * HW;  // [2,64,128,128]
    const size_t IMASK_OFF = CMASK_OFF + (size_t)B_SZ * LATC * HW;  // [2,4,64,128,128]
    const size_t CENT_OFF  = IMASK_OFF + (size_t)NIMG * LATC * HW;
    const size_t IENT_OFF  = CENT_OFF  + (size_t)B_SZ * LATC * HW;
    const size_t CMEAN_OFF = IENT_OFF  + (size_t)NIMG * LATC * HW;

    // ---- workspace carve (byte-based, 256B aligned) ----
    char* base = (char*)d_ws;
    size_t off = 0;
    auto allocB = [&](size_t bytes) { void* p = base + off; off += (bytes + 255) & ~(size_t)255; return p; };
    bfu* bufA   = (bfu*)allocB((size_t)NIMG * INCH * HW * 2);
    bfu* bufB   = (bfu*)allocB((size_t)NIMG * INCH * HW * 2);
    bfu* bufC   = (bfu*)allocB((size_t)NIMG * INCH * HW * 2);
    bfu* s0     = (bfu*)allocB((size_t)NIMG * MIDC * HW * 2);
    bfu* z      = (bfu*)allocB((size_t)NIMG * LATC * HW * 2);
    bfu* ind    = (bfu*)allocB((size_t)NIMG * LATC * HW * 2);
    bfu* com    = (bfu*)allocB((size_t)B_SZ * LATC * HW * 2);
    bfu* tc     = (bfu*)allocB((size_t)B_SZ * LATC * HW * 2);
    bfu* zdec   = (bfu*)allocB((size_t)NIMG * LATC * HW * 2);
    bfu* t16    = (bfu*)allocB((size_t)NIMG * 16 * HW * 2);
    bfu* m1     = (bfu*)allocB((size_t)NIMG * 16 * HW * 2);
    bfu* m2     = (bfu*)allocB((size_t)NIMG * 16 * HW * 2);
    float* wts    = (float*)allocB((size_t)NIMG * HW * 4);
    float* sc     = (float*)allocB((size_t)NIMG * HW * 4);
    float* pooled = (float*)allocB((size_t)NIMG * LATC * 1024 * 4);
    float* sepool = (float*)allocB((size_t)NIMG * INCH * 4);
    float* sescl  = (float*)allocB((size_t)NIMG * INCH * 4);

    // ---- parameter extraction (insertion-order DFS of params dict) ----
    int pi = 1;
    #define NEXTP() ((const float*)d_in[pi++])
    #define NEXT_RES(nm) ResP nm; nm.gn_w=NEXTP(); nm.gn_b=NEXTP(); nm.w1=NEXTP(); nm.b1=NEXTP(); \
                         nm.wd=NEXTP(); nm.bd=NEXTP(); nm.w2=NEXTP(); nm.b2=NEXTP();
    #define NEXT_GATE(nm) GateP nm; nm.w1=NEXTP(); nm.b1=NEXTP(); nm.w3=NEXTP(); nm.b3=NEXTP(); \
                          nm.w2=NEXTP(); nm.b2=NEXTP();

    const float* d0_pw = NEXTP(); const float* d0_pb = NEXTP();
    NEXT_RES(d0r1); NEXT_RES(d0r2);
    const float* d0_se_w1=NEXTP(); const float* d0_se_b1=NEXTP();
    const float* d0_se_w2=NEXTP(); const float* d0_se_b2=NEXTP();

    const float* d1_pw = NEXTP(); const float* d1_pb = NEXTP();
    NEXT_RES(d1r1); NEXT_RES(d1r2);
    const float* d1_se_w1=NEXTP(); const float* d1_se_b1=NEXTP();
    const float* d1_se_w2=NEXTP(); const float* d1_se_b2=NEXTP();

    NEXT_RES(bnres);

    const float* u0_pw = NEXTP(); const float* u0_pb = NEXTP();
    NEXT_RES(u0r1); NEXT_RES(u0r2);
    const float* u1_w = NEXTP(); const float* u1_b = NEXTP();
    NEXT_RES(u1r1); NEXT_RES(u1r2);

    const float* tw1 = NEXTP(); const float* tb1 = NEXTP();
    const float* tw2 = NEXTP(); const float* tb2 = NEXTP();

    NEXT_GATE(cgate); NEXT_GATE(igate);

    // ================= encoder =================
    // down0: 128 -> 96 (fp32 network input -> bf16 resident)
    launch_conv(feats, INCH, nullptr, 0, d0_pw, d0_pb, nullptr, bufA, MIDC, NIMG, 0, true, false, stream);
    run_res(d0r1, bufA, bufB, bufC, MIDC, NIMG, stream);
    run_res(d0r2, bufA, bufB, bufC, MIDC, NIMG, stream);
    run_se(bufA, d0_se_w1, d0_se_b1, d0_se_w2, d0_se_b2, s0, sepool, sescl, MIDC, 24, stream);

    // down1: 96 -> 64
    launch_conv(s0, MIDC, nullptr, 0, d1_pw, d1_pb, nullptr, bufA, LATC, NIMG, 0, false, false, stream);
    run_res(d1r1, bufA, bufB, bufC, LATC, NIMG, stream);
    run_res(d1r2, bufA, bufB, bufC, LATC, NIMG, stream);
    run_se(bufA, d1_se_w1, d1_se_b1, d1_se_w2, d1_se_b2, z, sepool, sescl, LATC, 16, stream);

    // bottleneck
    run_res(bnres, z, bufB, bufC, LATC, NIMG, stream);

    // ================= temporal decomposition =================
    launch_conv(z, LATC, nullptr, 0, tw1, tb1, nullptr, t16, 16, NIMG, FLAG_GELU_OUT, false, false, stream);
    launch_conv(t16, 16, nullptr, 0, tw2, tb2, nullptr, sc, 1, NIMG, 0, false, true, stream);
    softmax_t_kernel<<<(unsigned)((size_t)B_SZ * HW / 256), 256, 0, stream>>>(sc, wts);
    combine_kernel<<<(unsigned)((size_t)B_SZ * LATC * HW / 256), 256, 0, stream>>>(z, wts, com, ind);

    // ================= gates =================
    run_gate(cgate, com, B_SZ, 256, m1, m2, out + CENT_OFF, pooled,
             out + CMASK_OFF, tc, stream);                      // tc = common * cmask
    run_gate(igate, ind, NIMG, 102, m1, m2, out + IENT_OFF, pooled,
             out + IMASK_OFF, ind, stream);                     // ti = ind * imask (in place)

    zdec_kernel<<<(unsigned)((size_t)NIMG * LATC * HW / 256), 256, 0, stream>>>(tc, ind, zdec);

    // ================= decoder =================
    // up0: concat(zdec[64], s0[96]) -> 96, single pass (c1=64 is K-step aligned)
    launch_conv(zdec, LATC, s0, MIDC, u0_pw, u0_pb, nullptr, bufA, MIDC, NIMG, 0, false, false, stream);
    run_res(u0r1, bufA, bufB, bufC, MIDC, NIMG, stream);
    run_res(u0r2, bufA, bufB, bufC, MIDC, NIMG, stream);
    // up1: 96 -> 128
    launch_conv(bufA, MIDC, nullptr, 0, u1_w, u1_b, nullptr, bufC, INCH, NIMG, 0, false, false, stream);
    run_res(u1r1, bufC, bufA, bufB, INCH, NIMG, stream);
    run_res(u1r2, bufC, bufA, bufB, INCH, NIMG, stream);

    cvt_out_kernel<<<(unsigned)((size_t)NIMG * INCH * HW / 256), 256, 0, stream>>>(bufC, out + OUT_OFF);

    // mask means are exact: keep/1024 blocks kept, 16 px per block over 16384 px
    set_means_kernel<<<1, 1, 0, stream>>>(out + CMEAN_OFF, 256.f / 1024.f, 102.f / 1024.f);
}
